// AttentionBlock_4930622455928
// MI455X (gfx1250) — matrix operand order repaired and checked by reading, not compiled, not run on hardware
//
#include <hip/hip_runtime.h>
#include <hip/hip_bf16.h>

// ---------------------------------------------------------------------------
// Causal MHA block for MI455X (gfx1250, wave32, WMMA).
// f16 WMMA (f32 accumulate); single-pass flash attention (logits never hit
// HBM); 32x64 register tiles in the GEMMs for A/B fragment reuse; V stored
// transposed so the P@V B-fragment is a contiguous-K vector load.
// ---------------------------------------------------------------------------

typedef __attribute__((ext_vector_type(16))) _Float16 v16h;
typedef __attribute__((ext_vector_type(8)))  _Float16 v8h;
typedef __attribute__((ext_vector_type(8)))  float    v8f;

#define WMMA_F16(a, b, c) \
  __builtin_amdgcn_wmma_f32_16x16x32_f16(false, (a), false, (b), (short)0, (c), false, false)

// A-matrix (16x32 f16) per-lane fragment: m = lane&15, kb = 8*(lane>>4),
// elements = [kb..kb+7] then [16+kb..16+kb+7]  -> two 16B loads.
static __device__ __forceinline__ v16h load_a_f16(const _Float16* p) {
  v8h lo = *(const v8h*)(p);
  v8h hi = *(const v8h*)(p + 16);
  return __builtin_shufflevector(lo, hi, 0,1,2,3,4,5,6,7,8,9,10,11,12,13,14,15);
}
// B-matrix (32x16 f16) per-lane fragment: n = lane&15, kb = 8*(lane>>4),
// elements = [kb..kb+7] then [16+kb..16+kb+7] -> two 16B loads.
static __device__ __forceinline__ v16h load_b_f16(const _Float16* p) {
  v8h lo = *(const v8h*)(p);
  v8h hi = *(const v8h*)(p + 16);
  return __builtin_shufflevector(lo, hi, 0,1,2,3,4,5,6,7,8,9,10,11,12,13,14,15);
}

static __device__ __forceinline__ float halfmax16(float v) {
  #pragma unroll
  for (int off = 8; off; off >>= 1) v = fmaxf(v, __shfl_xor(v, off, 16));
  return v;
}
static __device__ __forceinline__ float halfsum16(float v) {
  #pragma unroll
  for (int off = 8; off; off >>= 1) v += __shfl_xor(v, off, 16);
  return v;
}

// ----------------------------- precision prep ------------------------------

__global__ void cvt_f32_to_f16(const float* __restrict__ src,
                               _Float16* __restrict__ dst, int n) {
  int i = blockIdx.x * blockDim.x + threadIdx.x;
  if (i < n) dst[i] = (_Float16)src[i];
}

// W is [K][N] row-major; WT is [N][K] so B fragments are contiguous-K loads.
__global__ void transpose_w_f16(const float* __restrict__ W,
                                _Float16* __restrict__ WT, int K, int N) {
  int o = blockIdx.x * blockDim.x + threadIdx.x;
  if (o < K * N) {
    int n = o / K, k = o % K;
    WT[o] = (_Float16)W[(size_t)k * N + n];
  }
}

// ------------------------------- QKV GEMM ----------------------------------
// qkv[8192,1536] = x[8192,512] @ W_in[512,1536] + b_in.
// One wave per 32x64 output tile: 2 A-frags x 4 B-frags -> 8 WMMAs / k-chunk.
// q,k scattered to [B,H,L,64]; v scattered TRANSPOSED to [B,H,64,L].

__global__ void __launch_bounds__(128)
gemm_qkv(const _Float16* __restrict__ xh, const _Float16* __restrict__ winT,
         const float* __restrict__ b_in,
         _Float16* __restrict__ qh, _Float16* __restrict__ kh,
         _Float16* __restrict__ vT) {
  const int wave = threadIdx.x >> 5, lane = threadIdx.x & 31;
  const int tile = blockIdx.x * 4 + wave;          // 256 * 24 tiles
  const int mt = tile % 256, nt = tile / 256;
  const int m0 = mt * 32, n0 = nt * 64;
  const int ln = lane & 15;
  const int kbA = (lane >> 4) * 8;
  const int kbB = (lane >> 4) * 8;
  const int half8 = (lane >> 4) * 8;

  const _Float16* arow0 = xh + (size_t)(m0 + ln) * 512;
  const _Float16* arow1 = xh + (size_t)(m0 + 16 + ln) * 512;

  v8f c[2][4] = {};
  #pragma unroll
  for (int k0 = 0; k0 < 512; k0 += 32) {
    v16h a0 = load_a_f16(arow0 + k0 + kbA);
    v16h a1 = load_a_f16(arow1 + k0 + kbA);
    #pragma unroll
    for (int ni = 0; ni < 4; ++ni) {
      v16h b = load_b_f16(winT + (size_t)(n0 + 16 * ni + ln) * 512 + k0 + kbB);
      c[0][ni] = WMMA_F16(a0, b, c[0][ni]);
      c[1][ni] = WMMA_F16(a1, b, c[1][ni]);
    }
  }

  #pragma unroll
  for (int ni = 0; ni < 4; ++ni) {
    const int col = n0 + 16 * ni + ln;       // 0..1535 (h,t uniform per ni)
    const int h = col / 192, rem = col % 192;
    const int t = rem / 64,  d = rem % 64;   // t: 0=q 1=k 2=v
    const float bias = b_in[col];
    #pragma unroll
    for (int mi = 0; mi < 2; ++mi) {
      #pragma unroll
      for (int r = 0; r < 8; ++r) {
        const int row = m0 + 16 * mi + r + half8;     // 0..8191
        const int b_ = row >> 11, l_ = row & 2047;
        const _Float16 val = (_Float16)(c[mi][ni][r] + bias);
        if (t == 2)       // V transposed: [bh][d][l]
          vT[((size_t)(b_ * 8 + h) * 64 + d) * 2048 + l_] = val;
        else {            // Q/K row-major: [bh][l][d]
          _Float16* dst = (t == 0) ? qh : kh;
          dst[((size_t)(b_ * 8 + h) * 2048 + l_) * 64 + d] = val;
        }
      }
    }
  }
}

// ---------------------------- flash attention ------------------------------
// One wave per (b,h, 16-row q-tile). Streams causal KV in 32-wide tiles:
// S (2 C-tiles, 4 WMMAs), online softmax, P re-layout via per-wave LDS,
// O += P@V (4 WMMAs, vectorized B loads from transposed V).

__global__ void __launch_bounds__(128)
flash_attn(const _Float16* __restrict__ qh, const _Float16* __restrict__ kh,
           const _Float16* __restrict__ vT, _Float16* __restrict__ ah) {
  __shared__ _Float16 plds[4][16 * 32];
  const int wave = threadIdx.x >> 5, lane = threadIdx.x & 31;
  const int task = blockIdx.x * 4 + wave;          // 32*128 tasks
  const int qt = task & 127, bh = task >> 7;
  const int q0 = qt * 16;
  const _Float16* Q = qh + (size_t)bh * 2048 * 64;
  const _Float16* K = kh + (size_t)bh * 2048 * 64;
  const _Float16* V = vT + (size_t)bh * 64 * 2048; // [64][2048]

  const int ln = lane & 15;
  const int half = lane >> 4;
  const int kbA = half * 8;
  const int kbB = half * 8;

  const _Float16* qrow = Q + (size_t)(q0 + ln) * 64;
  const v16h aQ0 = load_a_f16(qrow + kbA);         // Dk 0..31
  const v16h aQ1 = load_a_f16(qrow + 32 + kbA);    // Dk 32..63

  v8f o0 = {}, o1 = {}, o2 = {}, o3 = {};
  float rM[8], rL[8];
  #pragma unroll
  for (int r = 0; r < 8; ++r) { rM[r] = -3.0e38f; rL[r] = 0.0f; }

  const float scale = 0.125f;                      // 1/sqrt(64)
  const int qEnd = q0 + 15;
  _Float16* P = plds[wave];

  for (int j0 = 0; j0 <= qEnd; j0 += 32) {
    // ---- S = (Q K^T) for 32 keys: two 16x16 C tiles ----
    v8f s0 = {}, s1 = {};
    {
      const _Float16* kr0 = K + (size_t)(j0 + ln) * 64;
      s0 = WMMA_F16(aQ0, load_b_f16(kr0 + kbB), s0);
      s0 = WMMA_F16(aQ1, load_b_f16(kr0 + 32 + kbB), s0);
      const _Float16* kr1 = K + (size_t)(j0 + 16 + ln) * 64;
      s1 = WMMA_F16(aQ0, load_b_f16(kr1 + kbB), s1);
      s1 = WMMA_F16(aQ1, load_b_f16(kr1 + 32 + kbB), s1);
    }

    // ---- causal mask + online softmax (row = cross-lane within 16) ----
    const int c0 = j0 + ln, c1 = c0 + 16;
    float p0[8], p1[8];
    #pragma unroll
    for (int r = 0; r < 8; ++r) {
      const int row = q0 + r + half * 8;
      float v0 = s0[r] * scale; if (c0 > row) v0 = -1.0e30f;
      float v1 = s1[r] * scale; if (c1 > row) v1 = -1.0e30f;
      const float mx = halfmax16(fmaxf(v0, v1));
      const float mN = fmaxf(rM[r], mx);
      const float alpha = __expf(rM[r] - mN);
      rM[r] = mN;
      const float e0 = __expf(v0 - mN);
      const float e1 = __expf(v1 - mN);
      rL[r] = rL[r] * alpha + halfsum16(e0 + e1);
      o0[r] *= alpha; o1[r] *= alpha; o2[r] *= alpha; o3[r] *= alpha;
      p0[r] = e0; p1[r] = e1;
    }

    // ---- re-layout P: C-layout regs -> LDS -> A-layout fragment ----
    #pragma unroll
    for (int r = 0; r < 8; ++r) {
      const int row = r + half * 8;
      P[row * 32 + ln]      = (_Float16)p0[r];
      P[row * 32 + 16 + ln] = (_Float16)p1[r];
    }
    asm volatile("s_wait_dscnt 0" ::: "memory");
    const v16h aP = load_a_f16(P + ln * 32 + kbA);

    // ---- O += P @ V  (B-frag = contiguous keys from transposed V) ----
    #pragma unroll
    for (int t = 0; t < 4; ++t) {
      const v16h bV = load_b_f16(V + (size_t)(t * 16 + ln) * 2048 + j0 + kbB);
      if (t == 0) o0 = WMMA_F16(aP, bV, o0);
      else if (t == 1) o1 = WMMA_F16(aP, bV, o1);
      else if (t == 2) o2 = WMMA_F16(aP, bV, o2);
      else o3 = WMMA_F16(aP, bV, o3);
    }
  }

  // ---- finalize: O / l, scatter heads into [M, 512] f16 ----
  const int b_ = bh >> 3, h_ = bh & 7;
  #pragma unroll
  for (int r = 0; r < 8; ++r) {
    const float inv = 1.0f / rL[r];
    const int row = q0 + r + half * 8;
    const size_t base = (size_t)(b_ * 2048 + row) * 512 + h_ * 64 + ln;
    ah[base + 0]  = (_Float16)(o0[r] * inv);
    ah[base + 16] = (_Float16)(o1[r] * inv);
    ah[base + 32] = (_Float16)(o2[r] * inv);
    ah[base + 48] = (_Float16)(o3[r] * inv);
  }
}

// --------------------------- output GEMM + residual ------------------------
// out[8192,512] = ah @ W_out + b_out + x; 32x64 tile per wave.

__global__ void __launch_bounds__(128)
gemm_out(const _Float16* __restrict__ ah, const _Float16* __restrict__ woutT,
         const float* __restrict__ b_out, const float* __restrict__ x,
         float* __restrict__ out) {
  const int wave = threadIdx.x >> 5, lane = threadIdx.x & 31;
  const int tile = blockIdx.x * 4 + wave;          // 256 * 8 tiles
  const int mt = tile % 256, nt = tile / 256;
  const int m0 = mt * 32, n0 = nt * 64;
  const int ln = lane & 15;
  const int kbA = (lane >> 4) * 8;
  const int kbB = (lane >> 4) * 8;
  const int half8 = (lane >> 4) * 8;

  const _Float16* arow0 = ah + (size_t)(m0 + ln) * 512;
  const _Float16* arow1 = ah + (size_t)(m0 + 16 + ln) * 512;

  v8f c[2][4] = {};
  #pragma unroll
  for (int k0 = 0; k0 < 512; k0 += 32) {
    v16h a0 = load_a_f16(arow0 + k0 + kbA);
    v16h a1 = load_a_f16(arow1 + k0 + kbA);
    #pragma unroll
    for (int ni = 0; ni < 4; ++ni) {
      v16h b = load_b_f16(woutT + (size_t)(n0 + 16 * ni + ln) * 512 + k0 + kbB);
      c[0][ni] = WMMA_F16(a0, b, c[0][ni]);
      c[1][ni] = WMMA_F16(a1, b, c[1][ni]);
    }
  }

  #pragma unroll
  for (int ni = 0; ni < 4; ++ni) {
    const int col = n0 + 16 * ni + ln;
    const float bias = b_out[col];
    #pragma unroll
    for (int mi = 0; mi < 2; ++mi) {
      #pragma unroll
      for (int r = 0; r < 8; ++r) {
        const int row = m0 + 16 * mi + r + half8;
        const size_t idx = (size_t)row * 512 + col;
        out[idx] = c[mi][ni][r] + bias + x[idx];
      }
    }
  }
}

// --------------------------------- launch ----------------------------------

extern "C" void kernel_launch(void* const* d_in, const int* in_sizes, int n_in,
                              void* d_out, int out_size, void* d_ws, size_t ws_size,
                              hipStream_t stream) {
  const float* x     = (const float*)d_in[0];   // [4,2048,512]
  const float* W_in  = (const float*)d_in[1];   // [512,1536]
  const float* b_in  = (const float*)d_in[2];   // [1536]
  const float* W_out = (const float*)d_in[3];   // [512,512]
  const float* b_out = (const float*)d_in[4];   // [512]
  float* out = (float*)d_out;                   // [4,2048,512]

  const int M = 8192, C = 512;
  const size_t N_X  = (size_t)M * C;            // 4,194,304
  const size_t N_WI = (size_t)C * 1536;         //   786,432
  const size_t N_WO = (size_t)C * 512;          //   262,144
  const size_t N_HD = (size_t)4 * 8 * 2048 * 64;// 4,194,304 per q/k/v

  _Float16* xh    = (_Float16*)d_ws;
  _Float16* winT  = xh    + N_X;
  _Float16* woutT = winT  + N_WI;
  _Float16* qh    = woutT + N_WO;
  _Float16* kh    = qh    + N_HD;
  _Float16* vT    = kh    + N_HD;               // [B,H,64,L] transposed
  _Float16* ah    = vT    + N_HD;               // [M,512] f16, total ~42 MB

  cvt_f32_to_f16<<<(int)((N_X + 255) / 256), 256, 0, stream>>>(x, xh, (int)N_X);
  transpose_w_f16<<<(int)((N_WI + 255) / 256), 256, 0, stream>>>(W_in, winT, 512, 1536);
  transpose_w_f16<<<(int)((N_WO + 255) / 256), 256, 0, stream>>>(W_out, woutT, 512, 512);

  gemm_qkv<<<1536, 128, 0, stream>>>(xh, winT, b_in, qh, kh, vT);    // 6144 waves
  flash_attn<<<1024, 128, 0, stream>>>(qh, kh, vT, ah);              // 4096 waves
  gemm_out<<<512, 128, 0, stream>>>(ah, woutT, b_out, x, out);       // 2048 waves
}
